// ModulatedDeformConvLSTMCell_73332271612364
// MI455X (gfx1250) — matrix operand order repaired and checked by reading, not compiled, not run on hardware
//
#include <hip/hip_runtime.h>

// ---------------------------------------------------------------------------
// ModulatedDeformConvLSTMCell for MI455X (gfx1250, wave32, WMMA).
//
// Decomposition per deformable conv:
//   off  = GEMM( ow[27x576]  , im2col(input)[576xN] ) + ob      (WMMA bf16)
//   smp  = bilinear-gather(input, off) * sigmoid(mask)  -> bf16 [N x 576]
//   gate+= GEMM( w[64x576]   , smp[576xN] )                     (WMMA bf16)
// N = B*H*W = 32768, K = Cin*9 = 576.  Panels stored column-major [n][k] in
// bf16 so each wave's B fragment (16 consecutive K values per lane) is one
// aligned 32-byte global load.  Weights are staged in LDS pre-swizzled into
// the A-fragment element order (stride 584 -> bank-conflict-free b128 reads).
// ---------------------------------------------------------------------------

typedef __attribute__((ext_vector_type(16))) __bf16 v16bf;
typedef __attribute__((ext_vector_type(8)))  __bf16 v8bf;
typedef __attribute__((ext_vector_type(8)))  float  v8f;

#define NTOT   32768          // B*H*W
#define KDIM   576            // Cin * 9
#define KSTRIDE 584           // LDS stride (bf16 elems): 584*2B/4 = 292 % 64 = 36 -> conflict-free
#define NB     128            // N columns per workgroup (8 waves x 16)

union FragA { v16bf v; v8bf h[2]; };

// ------------------------------ WMMA GEMM ----------------------------------
// C[m][n] (f32, row-major, ld = 32768)  =  A[m][k] (f32) * B[n][k] (bf16,
// column-major panel, ld = 576), m < M (<= MT16*16), optional bias + accum.
template <int MT16>
__global__ __launch_bounds__(256)
void gemm_wmma_kernel(const float* __restrict__ A,
                      const __bf16* __restrict__ Bm,
                      float* __restrict__ Cm,
                      const float* __restrict__ bias,
                      int M, int accumulate)
{
    extern __shared__ __bf16 sA[];          // [MT16*16][KSTRIDE], swizzled
    const int tid = threadIdx.x;

    // Stage A into LDS: f32 -> bf16, swizzled to A-fragment element order.
    // For k = blk*32 + w:  lane-half h = w>>4,  elem e = w&15
    for (int i = tid; i < MT16 * 16 * KDIM; i += 256) {
        int m = i / KDIM;
        int k = i - m * KDIM;
        int w = k & 31;
        int pos = (k & ~31) + w;
        float v = (m < M) ? A[i] : 0.0f;
        sA[m * KSTRIDE + pos] = (__bf16)v;
    }
    __syncthreads();

    const int wave = tid >> 5;
    const int lane = tid & 31;
    const int col  = lane & 15;
    const int hi   = lane >> 4;
    const long n   = (long)blockIdx.x * NB + wave * 16 + col;

    // B fragment source: 16 consecutive K at k = kb + hi*16, column n.
    const __bf16* bptr = Bm + n * (long)KDIM + hi * 16;
    const __bf16* aptr = sA + (lane & 15) * KSTRIDE + hi * 16;

    v8f acc[MT16];
#pragma unroll
    for (int mt = 0; mt < MT16; ++mt)
#pragma unroll
        for (int e = 0; e < 8; ++e) acc[mt][e] = 0.0f;

    for (int kb = 0; kb < KDIM; kb += 32) {
        v16bf bfrag = *(const v16bf*)(bptr + kb);           // 32B aligned
        if (kb + 32 < KDIM)
            __builtin_prefetch(bptr + kb + 32, 0, 1);       // global_prefetch_b8
#pragma unroll
        for (int mt = 0; mt < MT16; ++mt) {
            const __bf16* ap = aptr + mt * 16 * KSTRIDE + kb;
            FragA af;
            af.h[0] = *(const v8bf*)(ap);                   // ds_load_b128
            af.h[1] = *(const v8bf*)(ap + 8);               // ds_load_b128
            acc[mt] = __builtin_amdgcn_wmma_f32_16x16x32_bf16(
                false, af.v, false, bfrag, (short)0, acc[mt], false, false);
        }
    }

#pragma unroll
    for (int mt = 0; mt < MT16; ++mt) {
#pragma unroll
        for (int r = 0; r < 8; ++r) {
            int m = mt * 16 + hi * 8 + r;                   // C layout: VGPR r -> M=r / M=8+r
            if (m < M) {
                long idx = (long)m * NTOT + n;
                float v = acc[mt][r];
                if (bias) v += bias[m];
                if (accumulate) v += Cm[idx];
                Cm[idx] = v;
            }
        }
    }
}

// ------------------------------ im2col (bf16) ------------------------------
// Bm[n][cin*9 + ky*3 + kx] = X[b][cin][y+ky-1][x+kx-1] (zero padded)
__global__ __launch_bounds__(256)
void im2col_kernel(const float* __restrict__ X, __bf16* __restrict__ Bm)
{
    long t = (long)blockIdx.x * 256 + threadIdx.x;          // [0, 64*NTOT)
    int cin = (int)(t >> 15);
    int n   = (int)(t & (NTOT - 1));
    int b = n >> 12, yx = n & 4095, y = yx >> 6, x = yx & 63;
    const float* xp = X + ((long)(b * 64 + cin) << 12);
    __bf16* out = Bm + (long)n * KDIM + cin * 9;
#pragma unroll
    for (int ky = 0; ky < 3; ++ky) {
        int yy = y + ky - 1;
        bool yin = (unsigned)yy < 64u;
#pragma unroll
        for (int kx = 0; kx < 3; ++kx) {
            int xx = x + kx - 1;
            float v = (yin && (unsigned)xx < 64u) ? xp[yy * 64 + xx] : 0.0f;
            out[ky * 3 + kx] = (__bf16)v;
        }
    }
}

// --------------------- deformable sampling + modulation --------------------
// OFF is [27][NTOT] f32: ch 2k = dy_k, 2k+1 = dx_k, 18+k = mask logits.
__global__ __launch_bounds__(256)
void sample_kernel(const float* __restrict__ X, const float* __restrict__ OFF,
                   __bf16* __restrict__ S)
{
    long t = (long)blockIdx.x * 256 + threadIdx.x;          // [0, 9*NTOT)
    int k = (int)(t >> 15);
    int n = (int)(t & (NTOT - 1));
    int b = n >> 12, yx = n & 4095, y = yx >> 6, x = yx & 63;

    float dy = OFF[(long)(2 * k) * NTOT + n];
    float dx = OFF[(long)(2 * k + 1) * NTOT + n];
    float ml = OFF[(long)(18 + k) * NTOT + n];
    float mask = 1.0f / (1.0f + __expf(-ml));

    float py = (float)(y + k / 3 - 1) + dy;
    float px = (float)(x + k % 3 - 1) + dx;
    float y0f = floorf(py), x0f = floorf(px);
    float wy = py - y0f, wx = px - x0f;
    int y0 = (int)y0f, x0 = (int)x0f;
    int y1 = y0 + 1,  x1 = x0 + 1;
    bool iy0 = (unsigned)y0 < 64u, iy1 = (unsigned)y1 < 64u;
    bool ix0 = (unsigned)x0 < 64u, ix1 = (unsigned)x1 < 64u;
    int cy0 = min(max(y0, 0), 63), cy1 = min(max(y1, 0), 63);
    int cx0 = min(max(x0, 0), 63), cx1 = min(max(x1, 0), 63);
    float w00 = (1.0f - wy) * (1.0f - wx) * mask;
    float w01 = (1.0f - wy) * wx * mask;
    float w10 = wy * (1.0f - wx) * mask;
    float w11 = wy * wx * mask;
    if (!(iy0 && ix0)) w00 = 0.0f;
    if (!(iy0 && ix1)) w01 = 0.0f;
    if (!(iy1 && ix0)) w10 = 0.0f;
    if (!(iy1 && ix1)) w11 = 0.0f;
    int i00 = cy0 * 64 + cx0, i01 = cy0 * 64 + cx1;
    int i10 = cy1 * 64 + cx0, i11 = cy1 * 64 + cx1;

    const float* xb = X + ((long)b << 18);
    __bf16* sp = S + (long)n * KDIM + k;
    for (int cin = 0; cin < 64; ++cin) {
        const float* xc = xb + ((long)cin << 12);
        float v = w00 * xc[i00] + w01 * xc[i01] + w10 * xc[i10] + w11 * xc[i11];
        sp[cin * 9] = (__bf16)v;
    }
}

// ------------------------------- LSTM gates --------------------------------
__global__ __launch_bounds__(256)
void gate_kernel(const float* __restrict__ Gi, const float* __restrict__ Gf,
                 const float* __restrict__ Gc, const float* __restrict__ Go,
                 const float* __restrict__ C0,
                 const float* __restrict__ bi, const float* __restrict__ bfv,
                 const float* __restrict__ bc, const float* __restrict__ bo,
                 const float* __restrict__ wci, const float* __restrict__ wcf,
                 const float* __restrict__ wco,
                 float* __restrict__ out)
{
    long t = (long)blockIdx.x * 256 + threadIdx.x;          // [0, 64*NTOT)
    int m = (int)(t >> 15);
    int n = (int)(t & (NTOT - 1));
    int b = n >> 12, yx = n & 4095;
    long cidx = ((long)b << 18) + ((long)m << 12) + yx;     // [B,64,H,W]

    float c  = C0[cidx];
    float ig = 1.0f / (1.0f + __expf(-(Gi[t] + c * wci[m] + bi[m])));
    float fg = 1.0f / (1.0f + __expf(-(Gf[t] + c * wcf[m] + bfv[m])));
    float cn = fg * c + ig * tanhf(Gc[t] + bc[m]);
    float og = 1.0f / (1.0f + __expf(-(Go[t] + cn * wco[m] + bo[m])));
    float hn = og * tanhf(cn);

    out[cidx] = hn;                                         // h_new
    out[(1L << 21) + cidx] = cn;                            // c_new (offset 8*64*4096)
}

// ------------------------------- launcher ----------------------------------
extern "C" void kernel_launch(void* const* d_in, const int* in_sizes, int n_in,
                              void* d_out, int out_size, void* d_ws, size_t ws_size,
                              hipStream_t stream)
{
    (void)in_sizes; (void)n_in; (void)out_size; (void)ws_size;

    const float* x = (const float*)d_in[0];
    const float* h = (const float*)d_in[1];
    const float* c = (const float*)d_in[2];
    const float *W[8], *OW[8], *OB[8];
    for (int i = 0; i < 8; ++i) {
        W[i]  = (const float*)d_in[3 + 3 * i];
        OW[i] = (const float*)d_in[4 + 3 * i];
        OB[i] = (const float*)d_in[5 + 3 * i];
    }
    const float* bi  = (const float*)d_in[27];
    const float* bfv = (const float*)d_in[28];
    const float* bc  = (const float*)d_in[29];
    const float* bo  = (const float*)d_in[30];
    const float* wci = (const float*)d_in[31];
    const float* wcf = (const float*)d_in[32];
    const float* wco = (const float*)d_in[33];

    // Workspace layout (~113 MB total).
    char* ws = (char*)d_ws;
    size_t off = 0;
    __bf16* p_col = (__bf16*)(ws + off); off += (size_t)KDIM * NTOT * 2;   // 37.75 MB
    __bf16* p_smp = (__bf16*)(ws + off); off += (size_t)KDIM * NTOT * 2;   // 37.75 MB
    float*  p_off = (float*) (ws + off); off += (size_t)27 * NTOT * 4;     // 3.54 MB
    float*  p_g[4];
    for (int g = 0; g < 4; ++g) { p_g[g] = (float*)(ws + off); off += (size_t)64 * NTOT * 4; }

    const size_t ldsA2 = (size_t)2 * 16 * KSTRIDE * 2;   // 37,376 B
    const size_t ldsA4 = (size_t)4 * 16 * KSTRIDE * 2;   // 74,752 B (<< 320 KB/WGP)
    dim3 blk(256);

    // x-fed convs (1,3,5,7) share im2col(x) for their offset GEMMs.
    im2col_kernel<<<8192, blk, 0, stream>>>(x, p_col);
    for (int j = 0; j < 4; ++j) {
        int ci = 2 * j;
        gemm_wmma_kernel<2><<<NTOT / NB, blk, ldsA2, stream>>>(OW[ci], p_col, p_off, OB[ci], 27, 0);
        sample_kernel<<<9 * NTOT / 256, blk, 0, stream>>>(x, p_off, p_smp);
        gemm_wmma_kernel<4><<<NTOT / NB, blk, ldsA4, stream>>>(W[ci], p_smp, p_g[j], nullptr, 64, 0);
    }
    // h-fed convs (2,4,6,8), accumulating into the gate pre-activations.
    im2col_kernel<<<8192, blk, 0, stream>>>(h, p_col);
    for (int j = 0; j < 4; ++j) {
        int ci = 2 * j + 1;
        gemm_wmma_kernel<2><<<NTOT / NB, blk, ldsA2, stream>>>(OW[ci], p_col, p_off, OB[ci], 27, 0);
        sample_kernel<<<9 * NTOT / 256, blk, 0, stream>>>(h, p_off, p_smp);
        gemm_wmma_kernel<4><<<NTOT / NB, blk, ldsA4, stream>>>(W[ci], p_smp, p_g[j], nullptr, 64, 1);
    }

    gate_kernel<<<8192, blk, 0, stream>>>(p_g[0], p_g[1], p_g[2], p_g[3], c,
                                          bi, bfv, bc, bo, wci, wcf, wco,
                                          (float*)d_out);
}
